// MbValueLayer_11458972745983
// MI455X (gfx1250) — matrix operand order repaired and checked by reading, not compiled, not run on hardware
//
#include <hip/hip_runtime.h>
#include <stdint.h>

#define S_   512
#define K_   64
#define H_   8
#define B_   8
#define NMB_ 4
#define C_   17
#define NKC  (S_ / 32)              // 16 k-chunks of 32
#define CHUNK_DW (5 * 4 * 2 * 256)  // 10240 dwords = 40 KB per (b,h,kchunk)

typedef __attribute__((ext_vector_type(16))) __bf16 v16bf;
typedef __attribute__((ext_vector_type(8)))  float  v8f;

union BReg { v16bf v; unsigned int u[8]; };

#if defined(__has_builtin)
#if __has_builtin(__builtin_amdgcn_tensor_load_to_lds) && __has_builtin(__builtin_amdgcn_s_wait_tensorcnt)
#define HAS_TDM 1
#endif
#endif
#ifndef HAS_TDM
#define HAS_TDM 0
#endif

typedef unsigned int u32x4 __attribute__((ext_vector_type(4)));
typedef int          i32x8 __attribute__((ext_vector_type(8)));
typedef int          i32x4 __attribute__((ext_vector_type(4)));

// Issue one TDM DMA: contiguous CHUNK_DW dwords global -> LDS (wave-level op).
__device__ __forceinline__ void tdm_load_chunk(const unsigned int* gsrc, unsigned int lds_off) {
#if HAS_TDM
    unsigned long long ga = (unsigned long long)(uintptr_t)gsrc;
    u32x4 g0;
    g0.x = 1u;                                                    // count=1, user D#
    g0.y = lds_off;                                               // lds_addr (bytes)
    g0.z = (unsigned int)ga;                                      // global_addr[31:0]
    g0.w = (unsigned int)((ga >> 32) & 0x1FFFFFFu) | (2u << 30);  // global_addr[56:32] | type=2
    i32x8 g1;
    g1[0] = 0x00020000;                                  // data_size = 2 (4 bytes)
    g1[1] = (int)((CHUNK_DW & 0xFFFF) << 16);            // tensor_dim0[15:0]  (bits 63:48)
    g1[2] = (int)(((unsigned)CHUNK_DW >> 16) & 0xFFFF);  // tensor_dim0[31:16] (bits 79:64)
    g1[3] = (int)((CHUNK_DW & 0xFFFF) << 16);            // tile_dim0 (bits 127:112)
    g1[4] = 0;                                           // tile_dim1 = tile_dim2 = 0 (1-D)
    g1[5] = (int)CHUNK_DW;                               // tensor_dim0_stride[31:0]
    g1[6] = 0;
    g1[7] = 0;
    i32x4 z4 = {0, 0, 0, 0};
#if __clang_major__ >= 23
    i32x8 z8 = {0, 0, 0, 0, 0, 0, 0, 0};
    __builtin_amdgcn_tensor_load_to_lds(g0, g1, z4, z4, z8, 0);
#else
    __builtin_amdgcn_tensor_load_to_lds(g0, g1, z4, z4, 0);
#endif
#else
    (void)gsrc; (void)lds_off;
#endif
}

// Split: hi = x truncated to bf16, lo = (x - hi) truncated to bf16; x ~= hi + lo.
__device__ __forceinline__ void splitbf(float x, unsigned int& hi, unsigned int& lo) {
    unsigned int xb = __float_as_uint(x);
    hi = xb >> 16;
    float hf = __uint_as_float(xb & 0xffff0000u);
    lo = __float_as_uint(x - hf) >> 16;
}

// ---------------- Kernel 1: W2_[c,h,d,D] = sum_m softmax(alpha2[c,:,h])[m] * W2[m,h,d,D]
__global__ __launch_bounds__(256) void mix_w2_kernel(const float* __restrict__ W2,
                                                     const float* __restrict__ alpha2,
                                                     float* __restrict__ W2m) {
    int c = blockIdx.x >> 3;
    int h = blockIdx.x & 7;
    float w[NMB_];
    float mx = -3.4e38f;
#pragma unroll
    for (int m = 0; m < NMB_; ++m) { w[m] = alpha2[(c * NMB_ + m) * H_ + h]; mx = fmaxf(mx, w[m]); }
    float s = 0.f;
#pragma unroll
    for (int m = 0; m < NMB_; ++m) { w[m] = __expf(w[m] - mx); s += w[m]; }
    float inv = 1.0f / s;
    for (int idx = threadIdx.x; idx < K_ * K_; idx += 256) {
        float a = 0.f;
#pragma unroll
        for (int m = 0; m < NMB_; ++m) a += w[m] * inv * W2[(m * H_ + h) * K_ * K_ + idx];
        W2m[(c * H_ + h) * K_ * K_ + idx] = a;
    }
}

// ---------------- Kernel 2: build WMMA B-operand register images of routed U.
// U_r[b,h,j,D] = sum_d V[b,h,j,d] * W2m[c(r, b_seq[b,j]), h, d, D]
// Bimg layout (dwords): [bh][kc][r*4+dsub][plane(hi=0,lo=1)][lane*8+t]
__global__ __launch_bounds__(256) void build_bimg_kernel(const float* __restrict__ V,
                                                         const int* __restrict__ bseq,
                                                         const float* __restrict__ W2m,
                                                         unsigned int* __restrict__ Bimg) {
    int blk = blockIdx.x;
    int ds  = blk & 3;
    int t2  = blk >> 2;
    int r   = t2 % 5;  t2 /= 5;
    int kc  = t2 & 15;
    int bh  = t2 >> 4;
    int b   = bh >> 3, h = bh & 7;

    int lane = threadIdx.x >> 3;      // 0..31
    int t    = threadIdx.x & 7;       // 0..7
    int j0   = kc * 32 + ((lane >> 4) << 3) + 2 * t + ((t >> 2) << 3);
    int col  = ds * 16 + (lane & 15); // D in 0..63

    float u[2];
#pragma unroll
    for (int sjj = 0; sjj < 2; ++sjj) {
        int j = j0 + sjj;
        int q = bseq[b * S_ + j];
        int c = (r == 0 || q == 0) ? 0 : (r - 1) * 4 + q;
        const float* vrow = V + ((size_t)bh * S_ + j) * K_;
        const float* wcol = W2m + ((size_t)c * H_ + h) * K_ * K_ + col;
        float a = 0.f;
#pragma unroll 8
        for (int d = 0; d < K_; ++d) a += vrow[d] * wcol[(size_t)d * K_];
        u[sjj] = a;
    }
    unsigned int h0, l0, h1, l1;
    splitbf(u[0], h0, l0);
    splitbf(u[1], h1, l1);
    size_t base = ((((size_t)bh * NKC + kc) * 5 + r) * 4 + ds) * 512;
    Bimg[base + 0 * 256 + lane * 8 + t] = h0 | (h1 << 16);
    Bimg[base + 1 * 256 + lane * 8 + t] = l0 | (l1 << 16);
}

// ---------------- Kernel 3: out[b,h,i,:] = P[b,h,i,:] @ U_{b_seq[b,i]}
// TDM double-buffered LDS staging + bf16 hi/lo split WMMA.
__global__ __launch_bounds__(256) void mb_value_main_kernel(const float* __restrict__ P,
                                                            const int* __restrict__ bseq,
                                                            const unsigned int* __restrict__ Bimg,
                                                            float* __restrict__ out) {
#if HAS_TDM
    __shared__ unsigned int ldsb[2][CHUNK_DW];   // 80 KB, double buffered
#else
    __shared__ unsigned int ldsb[1][CHUNK_DW];   // 40 KB, single buffer
#endif

    int wg   = blockIdx.x;          // B*H*4 = 256
    int ib   = wg & 3;              // 128-row block
    int bh   = wg >> 2;             // b*H + h
    int b    = bh >> 3;
    int wave = threadIdx.x >> 5;    // 0..7 -> i-tile
    int lane = threadIdx.x & 31;
    int i0   = ib * 128 + wave * 16;

    v8f acc[5][4];
#pragma unroll
    for (int r = 0; r < 5; ++r)
#pragma unroll
        for (int d = 0; d < 4; ++d) acc[r][d] = (v8f){0.f,0.f,0.f,0.f,0.f,0.f,0.f,0.f};

    const unsigned int* bsrc = Bimg + (size_t)bh * NKC * CHUNK_DW;
    // A 16x32 bf16 lane layout: row = i0 + (lane&15); K halves selected by lane>>4.
    const float* pr = P + ((size_t)bh * S_ + (i0 + (lane & 15))) * S_;
    int sel = (lane >> 4) * 8;

#if HAS_TDM
    // Prologue: DMA chunk 0 into buffer 0.
    if (wave == 0) {
        tdm_load_chunk(bsrc, (unsigned int)(uintptr_t)&ldsb[0][0]);
        __builtin_amdgcn_s_wait_tensorcnt(0);
    }
    __syncthreads();
#endif

#pragma unroll 1
    for (int kc = 0; kc < NKC; ++kc) {
#if HAS_TDM
        // Kick DMA for the next chunk into the other buffer; overlaps with WMMAs below.
        if (wave == 0 && kc + 1 < NKC)
            tdm_load_chunk(bsrc + (size_t)(kc + 1) * CHUNK_DW,
                           (unsigned int)(uintptr_t)&ldsb[(kc + 1) & 1][0]);
        const unsigned int* lc = &ldsb[kc & 1][0];
#else
        {   // Synchronous fallback staging.
            const uint4* src4 = (const uint4*)(bsrc + (size_t)kc * CHUNK_DW);
            uint4*       dst4 = (uint4*)&ldsb[0][0];
#pragma unroll
            for (int s = 0; s < CHUNK_DW / 4 / 256; ++s)
                dst4[threadIdx.x + 256 * s] = src4[threadIdx.x + 256 * s];
        }
        __syncthreads();
        const unsigned int* lc = &ldsb[0][0];
#endif

        int k0 = kc * 32;
        if (kc + 1 < NKC) __builtin_prefetch(pr + k0 + 32, 0, 1);

        float4 f0 = *(const float4*)(pr + k0 + sel);
        float4 f1 = *(const float4*)(pr + k0 + sel + 4);
        float4 f2 = *(const float4*)(pr + k0 + 16 + sel);
        float4 f3 = *(const float4*)(pr + k0 + 16 + sel + 4);
        float fa[16] = { f0.x, f0.y, f0.z, f0.w, f1.x, f1.y, f1.z, f1.w,
                         f2.x, f2.y, f2.z, f2.w, f3.x, f3.y, f3.z, f3.w };
        BReg ahi, alo;
#pragma unroll
        for (int t = 0; t < 8; ++t) {
            unsigned int h0, l0, h1, l1;
            splitbf(fa[2 * t],     h0, l0);
            splitbf(fa[2 * t + 1], h1, l1);
            ahi.u[t] = h0 | (h1 << 16);
            alo.u[t] = l0 | (l1 << 16);
        }

#pragma unroll
        for (int r = 0; r < 5; ++r) {
#pragma unroll
            for (int d = 0; d < 4; ++d) {
                int off = (r * 4 + d) * 512 + lane * 8;
                BReg bhi, blo;
#pragma unroll
                for (int t = 0; t < 8; ++t) { bhi.u[t] = lc[off + t]; blo.u[t] = lc[off + 256 + t]; }
                acc[r][d] = __builtin_amdgcn_wmma_f32_16x16x32_bf16(
                    false, ahi.v, false, bhi.v, (short)0, acc[r][d], false, false);
                acc[r][d] = __builtin_amdgcn_wmma_f32_16x16x32_bf16(
                    false, ahi.v, false, blo.v, (short)0, acc[r][d], false, false);
                acc[r][d] = __builtin_amdgcn_wmma_f32_16x16x32_bf16(
                    false, alo.v, false, bhi.v, (short)0, acc[r][d], false, false);
            }
        }

#if HAS_TDM
        if (wave == 0) __builtin_amdgcn_s_wait_tensorcnt(0);  // next buffer filled
#endif
        __syncthreads();
    }

    // Epilogue: per-row class select among the 5 routed accumulators, write out.
    // C/D layout: VGPR v -> row M = v + (lane>=16 ? 8 : 0); col N = lane&15.
#pragma unroll
    for (int v = 0; v < 8; ++v) {
        int row = i0 + v + ((lane >> 4) << 3);
        int rc  = bseq[b * S_ + row];
#pragma unroll
        for (int d = 0; d < 4; ++d) {
            float val = acc[0][d][v];
            val = (rc == 1) ? acc[1][d][v] : val;
            val = (rc == 2) ? acc[2][d][v] : val;
            val = (rc == 3) ? acc[3][d][v] : val;
            val = (rc == 4) ? acc[4][d][v] : val;
            out[((size_t)bh * S_ + row) * K_ + d * 16 + (lane & 15)] = val;
        }
    }
}

extern "C" void kernel_launch(void* const* d_in, const int* in_sizes, int n_in,
                              void* d_out, int out_size, void* d_ws, size_t ws_size,
                              hipStream_t stream) {
    (void)in_sizes; (void)n_in; (void)out_size;
    const float* p_attn = (const float*)d_in[0];
    const float* value  = (const float*)d_in[1];
    const int*   b_seq  = (const int*)d_in[2];
    const float* W2     = (const float*)d_in[3];
    const float* alpha2 = (const float*)d_in[4];
    float* out = (float*)d_out;

    const size_t w2m_bytes  = (size_t)C_ * H_ * K_ * K_ * 4;        // 2,228,224
    const size_t bimg_bytes = (size_t)B_ * H_ * NKC * CHUNK_DW * 4; // 41,943,040
    if (ws_size < w2m_bytes + bimg_bytes) return;

    float*        W2m  = (float*)d_ws;
    unsigned int* Bimg = (unsigned int*)((char*)d_ws + w2m_bytes);

    mix_w2_kernel<<<C_ * H_, 256, 0, stream>>>(W2, alpha2, W2m);
    build_bimg_kernel<<<B_ * H_ * NKC * 5 * 4, 256, 0, stream>>>(value, b_seq, W2m, Bimg);
    mb_value_main_kernel<<<B_ * H_ * 4, 256, 0, stream>>>(p_attn, b_seq, Bimg, out);
}
